// Attention_24799141167824
// MI455X (gfx1250) — matrix operand order repaired and checked by reading, not compiled, not run on hardware
//
#include <hip/hip_runtime.h>
#include <stdint.h>

// ---------------- problem constants ----------------
#define BB     8
#define NN     1024
#define DIMC   256
#define HEADS  8
#define HD     32
#define TABLE  3969      // (2*32-1)^2
#define CPBH   512
#define QKVN   768

typedef float v8f   __attribute__((ext_vector_type(8)));
typedef __bf16 v16bf __attribute__((ext_vector_type(16)));
typedef unsigned int u32x4 __attribute__((ext_vector_type(4)));

union Frag {                 // one 16x32 (A) / 32x16 (B) bf16 WMMA operand
    v16bf v;
    u32x4 q[2];
    unsigned int u[8];
};

// ---------------- helpers ----------------
__device__ __forceinline__ unsigned short f2bf(float f) {    // RNE f32->bf16
    unsigned int u = __float_as_uint(f);
    u += 0x7FFFu + ((u >> 16) & 1u);
    return (unsigned short)(u >> 16);
}
__device__ __forceinline__ unsigned int packbf(float a, float b) {
    return (unsigned int)f2bf(a) | ((unsigned int)f2bf(b) << 16);
}
__device__ __forceinline__ float bflo(unsigned int u) { return __uint_as_float(u << 16); }
__device__ __forceinline__ float bfhi(unsigned int u) { return __uint_as_float(u & 0xFFFF0000u); }

__device__ __forceinline__ v8f wmma_bf16(v16bf a, v16bf b, v8f c) {
    return __builtin_amdgcn_wmma_f32_16x16x32_bf16(false, a, false, b, (short)0, c, false, false);
}

// ---------------- 1) f32 -> bf16 conversion ----------------
__global__ void cvt_bf16_kernel(const float* __restrict__ in,
                                unsigned short* __restrict__ out, int n) {
    int i = blockIdx.x * blockDim.x + threadIdx.x;
    if (i < n) out[i] = f2bf(in[i]);
}

// ---------------- 2) CPB MLP: table[h][t] ----------------
__global__ void cpb_kernel(const float* __restrict__ ct,
                           const float* __restrict__ w1, const float* __restrict__ b1,
                           const float* __restrict__ w2, const float* __restrict__ b2,
                           float* __restrict__ table) {
    __shared__ float red[HEADS][128];
    int row = blockIdx.x;
    int t   = threadIdx.x;
    float c0 = ct[row * 2 + 0], c1 = ct[row * 2 + 1];
    float acc[HEADS];
#pragma unroll
    for (int hh = 0; hh < HEADS; hh++) acc[hh] = 0.f;
    for (int j = t; j < CPBH; j += 128) {
        float hv = fmaxf(c0 * w1[j * 2] + c1 * w1[j * 2 + 1] + b1[j], 0.f);
#pragma unroll
        for (int hh = 0; hh < HEADS; hh++) acc[hh] += hv * w2[hh * CPBH + j];
    }
#pragma unroll
    for (int hh = 0; hh < HEADS; hh++) red[hh][t] = acc[hh];
    __syncthreads();
    for (int s = 64; s > 0; s >>= 1) {
        if (t < s) {
#pragma unroll
            for (int hh = 0; hh < HEADS; hh++) red[hh][t] += red[hh][t + s];
        }
        __syncthreads();
    }
    if (t < HEADS) table[t * TABLE + row] = red[t][0] + b2[t];
}

// ---------------- 3) expand rel-pos bias to bf16[H][N][N] ----------------
__global__ void bias_expand_kernel(const int* __restrict__ rpi,
                                   const float* __restrict__ table,
                                   unsigned short* __restrict__ bias) {
    int blk = blockIdx.x;
    int h = blk >> 10, n = blk & (NN - 1);
    const int* rrow = rpi + (size_t)n * NN;
    const float* th = table + h * TABLE;
    unsigned short* brow = bias + ((size_t)h * NN + n) * NN;
    for (int m = threadIdx.x; m < NN; m += blockDim.x) brow[m] = f2bf(th[rrow[m]]);
}

// ---------------- 4) bf16 WMMA GEMM: out = A @ W^T + bias ----------------
// 16(M) x 64(N) tile per wave: one A-fragment feeds 4 WMMAs per K-chunk.
// A: MxK bf16 row-major, W: NnxK bf16 row-major, out: MxNn f32; Nn % 64 == 0.
__global__ void __launch_bounds__(128)
gemm_bf16_kernel(const unsigned short* __restrict__ A,
                 const unsigned short* __restrict__ W,
                 const float* __restrict__ bias,
                 float* __restrict__ out, int M, int Nn, int K) {
    int lane = threadIdx.x & 31;
    int wid  = blockIdx.x * (blockDim.x >> 5) + (threadIdx.x >> 5);
    int ntiles = Nn >> 6;
    int mt = (wid / ntiles) << 4;
    int nt = (wid % ntiles) << 6;
    int lo = lane & 15, hf = lane >> 4;

    v8f acc0 = {0.f, 0.f, 0.f, 0.f, 0.f, 0.f, 0.f, 0.f};
    v8f acc1 = acc0, acc2 = acc0, acc3 = acc0;

    const unsigned short* arow = A + (size_t)(mt + lo) * K + hf * 16;  // A layout
    const unsigned short* w0   = W + (size_t)(nt + lo) * K + hf * 16;  // B layout
    const unsigned short* w1   = w0 + (size_t)16 * K;
    const unsigned short* w2   = w0 + (size_t)32 * K;
    const unsigned short* w3   = w0 + (size_t)48 * K;

    Frag a, b0, b1, b2, b3;
    for (int k0 = 0; k0 < K; k0 += 32) {
        a.q[0]  = *(const u32x4*)(arow + k0);
        a.q[1]  = *(const u32x4*)(arow + k0 + 8);
        b0.q[0] = *(const u32x4*)(w0 + k0);
        b0.q[1] = *(const u32x4*)(w0 + k0 + 8);
        b1.q[0] = *(const u32x4*)(w1 + k0);
        b1.q[1] = *(const u32x4*)(w1 + k0 + 8);
        b2.q[0] = *(const u32x4*)(w2 + k0);
        b2.q[1] = *(const u32x4*)(w2 + k0 + 8);
        b3.q[0] = *(const u32x4*)(w3 + k0);
        b3.q[1] = *(const u32x4*)(w3 + k0 + 8);
        acc0 = wmma_bf16(a.v, b0.v, acc0);
        acc1 = wmma_bf16(a.v, b1.v, acc1);
        acc2 = wmma_bf16(a.v, b2.v, acc2);
        acc3 = wmma_bf16(a.v, b3.v, acc3);
    }
    float* orow = out + (size_t)(mt + hf * 8) * Nn + nt + lo;
#pragma unroll
    for (int t = 0; t < 4; t++) {
        v8f acc = (t == 0) ? acc0 : (t == 1) ? acc1 : (t == 2) ? acc2 : acc3;
        float bn = bias[nt + t * 16 + lo];
#pragma unroll
        for (int r = 0; r < 8; r++) orow[(size_t)r * Nn + t * 16] = acc[r] + bn;
    }
}

// ---------------- 5) q/k/v prep: l2norm, softplus temp scale, V transpose ----------------
__global__ void qkvprep_kernel(const float* __restrict__ qkv,
                               const float* __restrict__ qemb,
                               const float* __restrict__ temp,
                               unsigned short* __restrict__ qs,
                               unsigned short* __restrict__ kn,
                               unsigned short* __restrict__ vt) {
    int warp = threadIdx.x >> 5, lane = threadIdx.x & 31;
    int wid = blockIdx.x * (blockDim.x >> 5) + warp;   // 0 .. B*H*N-1
    int b = wid >> 13;                                 // H*N = 8192
    int rem = wid & 8191;
    int h = rem >> 10, n = rem & 1023;
    const float* base = qkv + ((size_t)(b * NN + n)) * QKVN + h * HD;
    float qv = base[lane];
    float kv = base[256 + lane];
    float vv = base[512 + lane];
    float sq = qv * qv, sk = kv * kv;
#pragma unroll
    for (int s = 1; s < 32; s <<= 1) {
        sq += __shfl_xor(sq, s, 32);
        sk += __shfl_xor(sk, s, 32);
    }
    float qn = qv / fmaxf(sqrtf(sq), 1e-12f) + qemb[h * HD + lane];
    float scale = log1pf(__expf(temp[h])) * 6.9314718055994531f;   // softplus * ln(1024)
    float knv = kv / fmaxf(sqrtf(sk), 1e-12f);
    size_t ro = (((size_t)(b * HEADS + h)) * NN + n) * HD + lane;
    qs[ro] = f2bf(qn * scale);
    kn[ro] = f2bf(knv);
    vt[(((size_t)(b * HEADS + h)) * HD + lane) * NN + n] = f2bf(vv);
}

// ---------------- 6) fused flash attention (one wave per 16-query tile) ----------------
__global__ void __launch_bounds__(128)
attn_kernel(const unsigned short* __restrict__ qs,
            const unsigned short* __restrict__ kn,
            const unsigned short* __restrict__ vt,
            const unsigned short* __restrict__ bias,
            unsigned short* __restrict__ outb) {
    int warp = threadIdx.x >> 5, lane = threadIdx.x & 31;
    int wid = blockIdx.x * 4 + warp;        // 0 .. B*H*(N/16)-1 = 4095
    int b = wid >> 9;                       // H*(N/16) = 512
    int rem = wid & 511;
    int h = rem >> 6;
    int n0 = (rem & 63) << 4;
    int lo = lane & 15, hf = lane >> 4;

    const unsigned short* qbase = qs + ((size_t)(b * HEADS + h) * NN) * HD;
    const unsigned short* kbase = kn + ((size_t)(b * HEADS + h) * NN) * HD;
    const unsigned short* vbase = vt + ((size_t)(b * HEADS + h) * HD) * NN;
    const unsigned short* bbase = bias + ((size_t)(h * NN + n0 + lo)) * NN;  // this lane's query row

    // Q as B-fragment of S^T = Kn @ Qs^T  (lane = query column)
    Frag bq;
    {
        const unsigned short* qp = qbase + (size_t)(n0 + lo) * HD + hf * 8;
        bq.q[0] = *(const u32x4*)(qp);
        bq.q[1] = *(const u32x4*)(qp + 16);
    }

    v8f o0 = {0.f, 0.f, 0.f, 0.f, 0.f, 0.f, 0.f, 0.f};
    v8f o1 = o0;
    float mrow = -1e30f, lrow = 0.f;
    const v8f zero = {0.f, 0.f, 0.f, 0.f, 0.f, 0.f, 0.f, 0.f};
    int src = hf * 8;                       // bpermute base for row-stat broadcast

    for (int kc = 0; kc < NN; kc += 32) {
        // K chunk as two A-fragments (16 keys x 32 dims each)
        const unsigned short* kp0 = kbase + (size_t)(kc + lo) * HD + hf * 8;
        const unsigned short* vp0 = vbase + (size_t)lo * NN + kc + hf * 8;
        // prefetch next chunk's K / V / bias lines (speculative; dropped at tail)
        __builtin_prefetch(kp0 + 32 * HD, 0, 3);
        __builtin_prefetch(vp0 + 32, 0, 3);
        __builtin_prefetch(bbase + kc + 32 + hf * 8, 0, 3);

        Frag a0, a1;
        a0.q[0] = *(const u32x4*)(kp0);
        a0.q[1] = *(const u32x4*)(kp0 + 16);
        const unsigned short* kp1 = kp0 + 16 * HD;
        a1.q[0] = *(const u32x4*)(kp1);
        a1.q[1] = *(const u32x4*)(kp1 + 16);

        v8f t0 = wmma_bf16(a0.v, bq.v, zero);   // S^T keys kc..kc+15
        v8f t1 = wmma_bf16(a1.v, bq.v, zero);   // S^T keys kc+16..kc+31

        // relative-position bias (bf16, matches C-layout: lane q, reg r = key)
        u32x4 bw0 = *(const u32x4*)(bbase + kc + hf * 8);
        u32x4 bw1 = *(const u32x4*)(bbase + kc + 16 + hf * 8);
#pragma unroll
        for (int j = 0; j < 4; j++) {
            t0[2 * j]     += bflo(bw0[j]);
            t0[2 * j + 1] += bfhi(bw0[j]);
            t1[2 * j]     += bflo(bw1[j]);
            t1[2 * j + 1] += bfhi(bw1[j]);
        }

        // online softmax over 32 keys (per-lane regs + one xor-16 combine)
        float mloc = t0[0];
#pragma unroll
        for (int j = 1; j < 8; j++) mloc = fmaxf(mloc, t0[j]);
#pragma unroll
        for (int j = 0; j < 8; j++) mloc = fmaxf(mloc, t1[j]);
        mloc = fmaxf(mloc, __shfl_xor(mloc, 16, 32));
        float mnew  = fmaxf(mrow, mloc);
        float alpha = __expf(mrow - mnew);
        float sloc = 0.f;
#pragma unroll
        for (int j = 0; j < 8; j++) {
            t0[j] = __expf(t0[j] - mnew); sloc += t0[j];
            t1[j] = __expf(t1[j] - mnew); sloc += t1[j];
        }
        sloc += __shfl_xor(sloc, 16, 32);
        lrow = lrow * alpha + sloc;
        mrow = mnew;

        // P (16q x 32k) packs IN-LANE into A-fragment layout
        Frag pf;
        pf.u[0] = packbf(t0[0], t0[1]); pf.u[1] = packbf(t0[2], t0[3]);
        pf.u[2] = packbf(t0[4], t0[5]); pf.u[3] = packbf(t0[6], t0[7]);
        pf.u[4] = packbf(t1[0], t1[1]); pf.u[5] = packbf(t1[2], t1[3]);
        pf.u[6] = packbf(t1[4], t1[5]); pf.u[7] = packbf(t1[6], t1[7]);

        // V chunk as two B-fragments (32 keys x 16 dims each), from Vt (d-major)
        Frag vb0, vb1;
        vb0.q[0] = *(const u32x4*)(vp0);
        vb0.q[1] = *(const u32x4*)(vp0 + 16);
        const unsigned short* vp1 = vp0 + 16 * NN;
        vb1.q[0] = *(const u32x4*)(vp1);
        vb1.q[1] = *(const u32x4*)(vp1 + 16);

        // rescale accumulators: alpha[q] broadcast to C rows via bpermute
#pragma unroll
        for (int r = 0; r < 8; r++) {
            float ar = __shfl(alpha, src + r, 32);
            o0[r] *= ar;
            o1[r] *= ar;
        }
        o0 = wmma_bf16(pf.v, vb0.v, o0);   // dims 0..15
        o1 = wmma_bf16(pf.v, vb1.v, o1);   // dims 16..31
    }

    // epilogue: divide by row sums, store bf16 (b, n, h*32+d)
    unsigned short* obase = outb + ((size_t)(b * NN + n0)) * DIMC + h * HD;
#pragma unroll
    for (int r = 0; r < 8; r++) {
        float lr = __shfl(lrow, src + r, 32);
        float inv = 1.0f / lr;
        unsigned short* op = obase + (size_t)(r + hf * 8) * DIMC;
        op[lo]      = f2bf(o0[r] * inv);
        op[lo + 16] = f2bf(o1[r] * inv);
    }
}

// ---------------- launch ----------------
extern "C" void kernel_launch(void* const* d_in, const int* in_sizes, int n_in,
                              void* d_out, int out_size, void* d_ws, size_t ws_size,
                              hipStream_t stream) {
    const float* x        = (const float*)d_in[0];
    const int*   rpi      = (const int*)  d_in[3];
    const float* ctab     = (const float*)d_in[4];
    const float* qkv_w    = (const float*)d_in[5];
    const float* qkv_b    = (const float*)d_in[6];
    const float* qemb     = (const float*)d_in[7];
    const float* temp     = (const float*)d_in[8];
    const float* proj_w   = (const float*)d_in[9];
    const float* proj_b   = (const float*)d_in[10];
    const float* cpb1_w   = (const float*)d_in[11];
    const float* cpb1_b   = (const float*)d_in[12];
    const float* cpb2_w   = (const float*)d_in[13];
    const float* cpb2_b   = (const float*)d_in[14];

    char* ws = (char*)d_ws;
    size_t off = 0;
    auto alloc = [&](size_t bytes) {
        char* p = ws + off;
        off += (bytes + 255) & ~(size_t)255;
        return p;
    };
    const int nX = BB * NN * DIMC;                         // 2,097,152
    unsigned short* xb    = (unsigned short*)alloc((size_t)nX * 2);
    unsigned short* qkvwb = (unsigned short*)alloc((size_t)QKVN * DIMC * 2);
    unsigned short* projwb= (unsigned short*)alloc((size_t)DIMC * DIMC * 2);
    float*          qkvf  = (float*)alloc((size_t)BB * NN * QKVN * 4);
    unsigned short* qsb   = (unsigned short*)alloc((size_t)BB * HEADS * NN * HD * 2);
    unsigned short* knb   = (unsigned short*)alloc((size_t)BB * HEADS * NN * HD * 2);
    unsigned short* vtb   = (unsigned short*)alloc((size_t)BB * HEADS * HD * NN * 2);
    float*          btab  = (float*)alloc((size_t)HEADS * TABLE * 4);
    unsigned short* biasb = (unsigned short*)alloc((size_t)HEADS * NN * NN * 2);
    unsigned short* attnb = (unsigned short*)alloc((size_t)nX * 2);

    // 1) conversions to bf16
    cvt_bf16_kernel<<<nX / 256, 256, 0, stream>>>(x, xb, nX);
    cvt_bf16_kernel<<<(QKVN * DIMC) / 256, 256, 0, stream>>>(qkv_w, qkvwb, QKVN * DIMC);
    cvt_bf16_kernel<<<(DIMC * DIMC) / 256, 256, 0, stream>>>(proj_w, projwb, DIMC * DIMC);

    // 2) CPB MLP -> per-head bias table
    cpb_kernel<<<TABLE, 128, 0, stream>>>(ctab, cpb1_w, cpb1_b, cpb2_w, cpb2_b, btab);

    // 3) expand gathered bias to bf16 [H][N][N]
    bias_expand_kernel<<<HEADS * NN, 256, 0, stream>>>(rpi, btab, biasb);

    // 4) QKV GEMM: (8192x256) x (768x256)^T -> f32
    {
        int waves = (BB * NN / 16) * (QKVN / 64);          // 6144
        gemm_bf16_kernel<<<waves / 4, 128, 0, stream>>>(xb, qkvwb, qkv_b, qkvf,
                                                        BB * NN, QKVN, DIMC);
    }

    // 5) normalize q/k, scale q, transpose v
    qkvprep_kernel<<<(BB * HEADS * NN) / 8, 256, 0, stream>>>(qkvf, qemb, temp, qsb, knb, vtb);

    // 6) fused attention
    attn_kernel<<<(BB * HEADS * (NN / 16)) / 4, 128, 0, stream>>>(qsb, knb, vtb, biasb, attnb);

    // 7) output projection -> d_out (f32)
    {
        int waves = (BB * NN / 16) * (DIMC / 64);          // 2048
        gemm_bf16_kernel<<<waves / 4, 128, 0, stream>>>(attnb, projwb, proj_b, (float*)d_out,
                                                        BB * NN, DIMC, DIMC);
    }
}
